// CausalSelfAttention_84018150244414
// MI455X (gfx1250) — matrix operand order repaired and checked by reading, not compiled, not run on hardware
//
#include <hip/hip_runtime.h>
#include <hip/hip_bf16.h>

typedef __bf16 bf16_t;
typedef __attribute__((ext_vector_type(16))) __bf16 v16bf;
typedef __attribute__((ext_vector_type(8)))  float  v8f;
typedef __attribute__((ext_vector_type(4)))  unsigned int v4u;

#define WMMA_BF16(a, b, c) \
  __builtin_amdgcn_wmma_f32_16x16x32_bf16(false, (a), false, (b), (short)0, (c), false, false)

// Async global->LDS copy (CDNA5 TDM-lite path, tracked by ASYNCcnt).
// vdst = LDS byte address VGPR, vaddr = 64-bit global address.
#define ASYNC_COPY_B128(ldsptr, gptr)                                   \
  asm volatile("global_load_async_to_lds_b128 %0, %1, off"              \
               :: "v"((unsigned)(size_t)(ldsptr)), "v"(gptr) : "memory")
#define WAIT_ASYNC() asm volatile("s_wait_asynccnt 0" ::: "memory")

struct V16Pair { v4u lo; v4u hi; };

// Build a v16bf fragment from two 16-byte chunks (two ds/global b128 loads).
__device__ __forceinline__ v16bf frag2x128(const bf16_t* p0, const bf16_t* p1) {
  V16Pair u;
  u.lo = *(const v4u*)p0;
  u.hi = *(const v4u*)p1;
  return __builtin_bit_cast(v16bf, u);
}

// ---------------------------------------------------------------------------
// CDNA5 WMMA fragment loaders (wave32), ISA 7.12.2 layouts.
// ---------------------------------------------------------------------------
__device__ __forceinline__ v16bf load_a16(const bf16_t* base, int ld, int koff) {
  int lane = threadIdx.x & 31;
  int m = lane & 15, sel = lane >> 4;
  const bf16_t* row = base + m * ld + koff + sel * 8;
  return frag2x128(row, row + 16);
}

__device__ __forceinline__ v16bf load_b16t(const bf16_t* base, int stride, int koff, int noff) {
  int lane = threadIdx.x & 31;
  int n = noff + (lane & 15), sel = lane >> 4;
  const bf16_t* row = base + n * stride + koff + sel * 8;
  return frag2x128(row, row + 16);
}

// ---------------------------------------------------------------------------
// fp32 -> bf16 conversion
// ---------------------------------------------------------------------------
__global__ __launch_bounds__(256) void cvt_f32_bf16(const float* __restrict__ in,
                                                    bf16_t* __restrict__ out, int n) {
  int i = blockIdx.x * 256 + threadIdx.x;
  if (i < n) out[i] = (bf16_t)in[i];
}

// ---------------------------------------------------------------------------
// QKV GEMM: [8192 x 1024] x [1024 x 3072] + bias -> Q,K,V in [B,H,T,D] bf16.
// Q pre-scaled by 1/sqrt(D). 128x128 tile, 8 waves x 16-row strips.
// Double-buffered LDS; A tile staged with async global->LDS copies,
// B tile staged transposed through VGPRs (N-major, stride 40).
// ---------------------------------------------------------------------------
#define BSTRIDE 40

__global__ __launch_bounds__(256) void qkv_gemm(const bf16_t* __restrict__ X,
                                                const bf16_t* __restrict__ W,
                                                const float* __restrict__ bias,
                                                bf16_t* __restrict__ Q,
                                                bf16_t* __restrict__ K,
                                                bf16_t* __restrict__ V) {
  __shared__ v4u As4[2][512];                 // 128 x 32 bf16, row-major
  __shared__ bf16_t Bst[2][128 * BSTRIDE];    // 128 n-rows x 32 k, padded

  const int n0 = blockIdx.x * 128;
  const int m0 = blockIdx.y * 128;
  const int tid = threadIdx.x;
  const int wave = tid >> 5;
  const int lane = tid & 31;
  const int ln = lane & 15, sel = lane >> 4;

  auto stage = [&](int bf, int k0) {
#pragma unroll
    for (int i = 0; i < 2; ++i) {
      int idx = tid + i * 256;
      int ra = idx >> 2, ca = (idx & 3) << 3;
      ASYNC_COPY_B128(&As4[bf][idx], X + (size_t)(m0 + ra) * 1024 + k0 + ca);
      int rb = idx >> 4, cb = (idx & 15) << 3;          // rb: k row, cb: n col
      v4u w = *(const v4u*)(W + (size_t)(k0 + rb) * 3072 + n0 + cb);
      const bf16_t* ww = (const bf16_t*)&w;
#pragma unroll
      for (int j = 0; j < 8; ++j) Bst[bf][(cb + j) * BSTRIDE + rb] = ww[j];
    }
  };

  v8f acc[8] = {};
  stage(0, 0);
  WAIT_ASYNC();
  __syncthreads();
  int buf = 0;
  for (int k0 = 0; k0 < 1024; k0 += 32) {
    if (k0 + 32 < 1024) stage(buf ^ 1, k0 + 32);
    v16bf a = load_a16((const bf16_t*)As4[buf] + wave * 16 * 32, 32, 0);
    v16bf bfr[8];
#pragma unroll
    for (int s = 0; s < 8; ++s) bfr[s] = load_b16t(Bst[buf], BSTRIDE, 0, s * 16);
#pragma unroll
    for (int s = 0; s < 8; ++s) acc[s] = WMMA_BF16(a, bfr[s], acc[s]);
    WAIT_ASYNC();
    __syncthreads();
    buf ^= 1;
  }

#pragma unroll
  for (int s = 0; s < 8; ++s) {
    int n = n0 + s * 16 + ln;
    int part = n >> 10, c = n & 1023, h = c >> 6, d = c & 63;
    bf16_t* dst = (part == 0) ? Q : ((part == 1) ? K : V);
    float scale = (part == 0) ? 0.125f : 1.0f;  // 1/sqrt(64) folded into Q
    float bv = bias[n];
#pragma unroll
    for (int r = 0; r < 8; ++r) {
      int m = m0 + wave * 16 + sel * 8 + r;
      int bb = m >> 11, t = m & 2047;
      float v = (acc[s][r] + bv) * scale;
      dst[(((size_t)(bb * 16 + h)) * 2048 + t) * 64 + d] = (bf16_t)v;
    }
  }
}

// ---------------------------------------------------------------------------
// Flash attention: block = (b, h, 128-query tile). 8 waves x 16 query rows.
// Double-buffered K/V chunks; K staged with async global->LDS (row-major,
// d-pairs contiguous), V staged transposed (d-major, stride 72).
// ---------------------------------------------------------------------------
#define VSTRIDE 72

__global__ __launch_bounds__(256) void flash_attn(const bf16_t* __restrict__ Q,
                                                  const bf16_t* __restrict__ K,
                                                  const bf16_t* __restrict__ V,
                                                  bf16_t* __restrict__ Y) {
  __shared__ v4u Ks4[2][512];                 // 64 x 64 bf16 row-major (j, d)
  __shared__ bf16_t Vst[2][64 * VSTRIDE];     // 64 d-rows x 64 j, padded
  __shared__ bf16_t Ps[8 * 16 * 64];          // per-wave 16x64 P strips

  const int q0 = blockIdx.x * 128;
  const int h  = blockIdx.y;
  const int b  = blockIdx.z;
  const int tid = threadIdx.x, wave = tid >> 5, lane = tid & 31;
  const int ln = lane & 15, sel = lane >> 4;

  const size_t headoff = ((size_t)(b * 16 + h)) * 2048;
  const bf16_t* Qbase = Q + (headoff + q0 + wave * 16) * 64;

  auto stage = [&](int bf, int k0) {
    const bf16_t* Kb = K + (headoff + k0) * 64;
    const bf16_t* Vb = V + (headoff + k0) * 64;
#pragma unroll
    for (int i = 0; i < 2; ++i) {
      int idx = tid + i * 256;
      int rr = idx >> 3, cc = (idx & 7) << 3;           // rr: key j, cc: d
      ASYNC_COPY_B128(&Ks4[bf][idx], Kb + rr * 64 + cc);
      v4u w = *(const v4u*)(Vb + rr * 64 + cc);
      const bf16_t* ww = (const bf16_t*)&w;
#pragma unroll
      for (int j = 0; j < 8; ++j) Vst[bf][(cc + j) * VSTRIDE + rr] = ww[j];
    }
  };

  // Q fragments in registers for the whole kernel (2 K-steps over D=64)
  v16bf aq[2];
#pragma unroll
  for (int st = 0; st < 2; ++st) {
    const bf16_t* row = Qbase + ln * 64 + st * 32 + sel * 8;
    aq[st] = frag2x128(row, row + 16);
  }

  v8f oacc[4] = {};
  float mstate[8], lstate[8];
#pragma unroll
  for (int r = 0; r < 8; ++r) { mstate[r] = -3.0e38f; lstate[r] = 0.0f; }

  bf16_t* Pw = Ps + wave * 16 * 64;
  const int trow = q0 + wave * 16 + sel * 8;
  const int nch = (q0 + 127) / 64 + 1;

  stage(0, 0);
  WAIT_ASYNC();
  __syncthreads();
  int buf = 0;
  for (int ci = 0; ci < nch; ++ci) {
    const int k0 = ci * 64;
    if (ci + 1 < nch) stage(buf ^ 1, k0 + 64);
    if (ci + 2 < nch) {  // L2 prefetch two chunks ahead (global_prefetch_b8)
      __builtin_prefetch(K + (headoff + k0 + 128) * 64, 0, 1);
      __builtin_prefetch(V + (headoff + k0 + 128) * 64, 0, 1);
    }
    const bf16_t* Ksp = (const bf16_t*)Ks4[buf];

    // S = Q (pre-scaled) @ K^T : preload all 8 B-frags, then WMMA burst
    v16bf bkf[8];
#pragma unroll
    for (int st = 0; st < 2; ++st)
#pragma unroll
      for (int sub = 0; sub < 4; ++sub) {
        const bf16_t* krow = Ksp + (sub * 16 + ln) * 64 + st * 32 + sel * 8;
        bkf[st * 4 + sub] = frag2x128(krow, krow + 16);
      }
    v8f sacc[4] = {};
#pragma unroll
    for (int st = 0; st < 2; ++st)
#pragma unroll
      for (int sub = 0; sub < 4; ++sub)
        sacc[sub] = WMMA_BF16(aq[st], bkf[st * 4 + sub], sacc[sub]);

    // causal mask + online softmax (row reductions within 16-lane halves)
#pragma unroll
    for (int r = 0; r < 8; ++r) {
      int row = trow + r;
      float rowm = -3.0e38f;
#pragma unroll
      for (int sub = 0; sub < 4; ++sub) {
        int j = k0 + sub * 16 + ln;
        float sv = sacc[sub][r];
        if (j > row) sv = -3.0e38f;
        sacc[sub][r] = sv;
        rowm = fmaxf(rowm, sv);
      }
#pragma unroll
      for (int off = 1; off < 16; off <<= 1)
        rowm = fmaxf(rowm, __shfl_xor(rowm, off, 32));
      float mnew = fmaxf(mstate[r], rowm);
      float alpha = __expf(mstate[r] - mnew);
      float rsum = 0.0f;
#pragma unroll
      for (int sub = 0; sub < 4; ++sub) {
        float p = __expf(sacc[sub][r] - mnew);
        sacc[sub][r] = p;
        rsum += p;
      }
#pragma unroll
      for (int off = 1; off < 16; off <<= 1)
        rsum += __shfl_xor(rsum, off, 32);
      lstate[r] = lstate[r] * alpha + rsum;
      mstate[r] = mnew;
#pragma unroll
      for (int sub = 0; sub < 4; ++sub) oacc[sub][r] *= alpha;
    }

    // P (C-layout regs) -> wave-private LDS strip, A-source order
#pragma unroll
    for (int sub = 0; sub < 4; ++sub)
#pragma unroll
      for (int r = 0; r < 8; ++r)
        Pw[(sel * 8 + r) * 64 + sub * 16 + ln] = (bf16_t)sacc[sub][r];

    // O += P @ V : preload A- and B-frags, then WMMA burst
    v16bf apf[2], bvf[8];
#pragma unroll
    for (int st = 0; st < 2; ++st) {
      apf[st] = load_a16(Pw, 64, st * 32);
#pragma unroll
      for (int sub = 0; sub < 4; ++sub)
        bvf[st * 4 + sub] = load_b16t(Vst[buf], VSTRIDE, st * 32, sub * 16);
    }
#pragma unroll
    for (int st = 0; st < 2; ++st)
#pragma unroll
      for (int sub = 0; sub < 4; ++sub)
        oacc[sub] = WMMA_BF16(apf[st], bvf[st * 4 + sub], oacc[sub]);

    WAIT_ASYNC();
    __syncthreads();
    buf ^= 1;
  }

  // normalize, write Y[b, t, h*64 + d] (bf16 input to proj GEMM)
#pragma unroll
  for (int r = 0; r < 8; ++r) {
    float inv = 1.0f / lstate[r];
    int t = trow + r;
#pragma unroll
    for (int sub = 0; sub < 4; ++sub) {
      float o = oacc[sub][r] * inv;
      Y[((size_t)b * 2048 + t) * 1024 + h * 64 + sub * 16 + ln] = (bf16_t)o;
    }
  }
}

// ---------------------------------------------------------------------------
// Proj GEMM: [8192 x 1024] x [1024 x 1024] + bias -> f32 output [B,T,C]
// ---------------------------------------------------------------------------
__global__ __launch_bounds__(256) void proj_gemm(const bf16_t* __restrict__ X,
                                                 const bf16_t* __restrict__ W,
                                                 const float* __restrict__ bias,
                                                 float* __restrict__ out) {
  __shared__ v4u As4[2][512];
  __shared__ bf16_t Bst[2][128 * BSTRIDE];

  const int n0 = blockIdx.x * 128;
  const int m0 = blockIdx.y * 128;
  const int tid = threadIdx.x;
  const int wave = tid >> 5;
  const int lane = tid & 31;
  const int ln = lane & 15, sel = lane >> 4;

  auto stage = [&](int bf, int k0) {
#pragma unroll
    for (int i = 0; i < 2; ++i) {
      int idx = tid + i * 256;
      int ra = idx >> 2, ca = (idx & 3) << 3;
      ASYNC_COPY_B128(&As4[bf][idx], X + (size_t)(m0 + ra) * 1024 + k0 + ca);
      int rb = idx >> 4, cb = (idx & 15) << 3;
      v4u w = *(const v4u*)(W + (size_t)(k0 + rb) * 1024 + n0 + cb);
      const bf16_t* ww = (const bf16_t*)&w;
#pragma unroll
      for (int j = 0; j < 8; ++j) Bst[bf][(cb + j) * BSTRIDE + rb] = ww[j];
    }
  };

  v8f acc[8] = {};
  stage(0, 0);
  WAIT_ASYNC();
  __syncthreads();
  int buf = 0;
  for (int k0 = 0; k0 < 1024; k0 += 32) {
    if (k0 + 32 < 1024) stage(buf ^ 1, k0 + 32);
    v16bf a = load_a16((const bf16_t*)As4[buf] + wave * 16 * 32, 32, 0);
    v16bf bfr[8];
#pragma unroll
    for (int s = 0; s < 8; ++s) bfr[s] = load_b16t(Bst[buf], BSTRIDE, 0, s * 16);
#pragma unroll
    for (int s = 0; s < 8; ++s) acc[s] = WMMA_BF16(a, bfr[s], acc[s]);
    WAIT_ASYNC();
    __syncthreads();
    buf ^= 1;
  }

#pragma unroll
  for (int s = 0; s < 8; ++s) {
    int n = n0 + s * 16 + ln;
    float bv = bias[n];
#pragma unroll
    for (int r = 0; r < 8; ++r) {
      int m = m0 + wave * 16 + sel * 8 + r;
      out[(size_t)m * 1024 + n] = acc[s][r] + bv;
    }
  }
}

// ---------------------------------------------------------------------------
// Host launch
// ---------------------------------------------------------------------------
extern "C" void kernel_launch(void* const* d_in, const int* in_sizes, int n_in,
                              void* d_out, int out_size, void* d_ws, size_t ws_size,
                              hipStream_t stream) {
  (void)in_sizes; (void)n_in; (void)out_size; (void)ws_size;
  const float* x      = (const float*)d_in[0];
  // d_in[1] = causal mask (bool) — applied analytically in flash_attn
  const float* W_attn = (const float*)d_in[2];
  const float* b_attn = (const float*)d_in[3];
  const float* W_proj = (const float*)d_in[4];
  const float* b_proj = (const float*)d_in[5];
  float* out = (float*)d_out;

  // workspace carve (bytes), all 16B-aligned
  char* ws = (char*)d_ws;
  bf16_t* xb  = (bf16_t*)(ws);              // 8388608 elems
  bf16_t* Wab = (bf16_t*)(ws + 16777216);   // 3145728 elems
  bf16_t* Wpb = (bf16_t*)(ws + 23068672);   // 1048576 elems
  bf16_t* Qw  = (bf16_t*)(ws + 25165824);   // 8388608 elems [B,H,T,D]
  bf16_t* Kw  = (bf16_t*)(ws + 41943040);
  bf16_t* Vw  = (bf16_t*)(ws + 58720256);
  bf16_t* Yb  = (bf16_t*)(ws + 75497472);   // 8388608 elems [B,T,C]

  cvt_f32_bf16<<<8388608 / 256, 256, 0, stream>>>(x, xb, 8388608);
  cvt_f32_bf16<<<3145728 / 256, 256, 0, stream>>>(W_attn, Wab, 3145728);
  cvt_f32_bf16<<<1048576 / 256, 256, 0, stream>>>(W_proj, Wpb, 1048576);

  qkv_gemm<<<dim3(24, 64), 256, 0, stream>>>(xb, Wab, b_attn, Qw, Kw, Vw);
  flash_attn<<<dim3(16, 16, 4), 256, 0, stream>>>(Qw, Kw, Vw, Yb);
  proj_gemm<<<dim3(8, 64), 256, 0, stream>>>(Yb, Wpb, b_proj, out);
}
